// RealNVP_16312285790404
// MI455X (gfx1250) — matrix operand order repaired and checked by reading, not compiled, not run on hardware
//
#include <hip/hip_runtime.h>

// ---------------------------------------------------------------------------
// RealNVP inverse pass, fused across all 6 coupling layers.
// MI455X / gfx1250: wave32, WMMA f32<-f16 16x16x32, weights staged in LDS.
// ---------------------------------------------------------------------------

typedef __attribute__((ext_vector_type(16))) _Float16 v16h;
typedef __attribute__((ext_vector_type(8)))  _Float16 v8h;
typedef __attribute__((ext_vector_type(8)))  float    v8f;

#define BN    262144      // B*N tokens
#define DD    64
#define HH    256
#define HALFD 32
#define LL    6

#define WAVES_PER_BLOCK 4
#define THREADS        (WAVES_PER_BLOCK * 32)
#define TOK_PER_WAVE   16
#define TOK_PER_BLOCK  (WAVES_PER_BLOCK * TOK_PER_WAVE)

// LDS row strides (pad to spread banks, keep 16B alignment of fragments)
#define W1R (HALFD + 8)   // 40 halves  (80 B rows)
#define W2R (HH + 8)      // 264 halves (528 B rows)
#define W3R (HH + 8)      // 264 halves
#define ZR  (DD + 4)      // 68 floats  (272 B rows)

// LDS layout (byte offsets; every region 16B aligned)
#define OFF_WRES   0
#define OFF_B1     (OFF_WRES + 32 * 4)
#define OFF_B2     (OFF_B1 + 256 * 4)
#define OFF_B3     (OFF_B2 + 256 * 4)
#define OFF_W1T    (OFF_B3 + 64 * 4)               // 2432
#define OFF_W2T    (OFF_W1T + 256 * W1R * 2)       // 22912
#define OFF_W3T    (OFF_W2T + 256 * W2R * 2)       // 158080
#define OFF_WAVE   (OFF_W3T + 64 * W3R * 2)        // 191872
#define WAVE_BYTES (2 * TOK_PER_WAVE * ZR * 4 + 2 * TOK_PER_WAVE * HH * 2)  // 25088
#define SMEM_BYTES (OFF_WAVE + WAVES_PER_BLOCK * WAVE_BYTES)   // 292224 (< 320 KB WGP LDS)

static __device__ inline v16h ld_frag(const _Float16* p0, const _Float16* p1) {
    union { v16h v; v8h h[2]; } u;
    u.h[0] = *(const v8h*)p0;   // 16B -> ds_load_b128
    u.h[1] = *(const v8h*)p1;   // 16B -> ds_load_b128
    return u.v;
}

static __device__ inline float lrelu(float v) {
    return fmaxf(v, 0.01f * v);   // v_mul + v_max, dual-issuable
}

__global__ __launch_bounds__(THREADS, 1)
void realnvp_fused(const float* __restrict__ x,
                   const float* __restrict__ W1, const float* __restrict__ b1,
                   const float* __restrict__ W2, const float* __restrict__ b2,
                   const float* __restrict__ W3, const float* __restrict__ b3,
                   const float* __restrict__ rv, const float* __restrict__ rg,
                   float* __restrict__ out)
{
    extern __shared__ char smem[];
    float*     wres = (float*)(smem + OFF_WRES);
    float*     b1s  = (float*)(smem + OFF_B1);
    float*     b2s  = (float*)(smem + OFF_B2);
    float*     b3s  = (float*)(smem + OFF_B3);
    _Float16*  W1T  = (_Float16*)(smem + OFF_W1T);
    _Float16*  W2T  = (_Float16*)(smem + OFF_W2T);
    _Float16*  W3T  = (_Float16*)(smem + OFF_W3T);

    const int wave = threadIdx.x >> 5;
    const int lane = threadIdx.x & 31;
    const int m    = lane & 15;     // row / col within 16-wide fragment
    const int hi   = lane >> 4;     // half-wave select

    char* wbase = smem + OFF_WAVE + wave * WAVE_BYTES;
    float*    zb0 = (float*)wbase;
    float*    zb1 = zb0 + TOK_PER_WAVE * ZR;
    _Float16* h1  = (_Float16*)(zb1 + TOK_PER_WAVE * ZR);
    _Float16* h2  = h1 + TOK_PER_WAVE * HH;

    const int waveTok = blockIdx.x * TOK_PER_BLOCK + wave * TOK_PER_WAVE;

    // w_res = g * v / ||v||  (block-shared, 32 values)
    if (threadIdx.x < 32) {
        float ss = 0.f;
        #pragma unroll
        for (int j = 0; j < HALFD; ++j) { float v = rv[j]; ss += v * v; }
        wres[threadIdx.x] = rg[0] * rv[threadIdx.x] * rsqrtf(ss);
    }

    // per-lane logdet partials: token r + hi*8, summed over this lane's columns
    float ldj_part[8];
    #pragma unroll
    for (int r = 0; r < 8; ++r) ldj_part[r] = 0.f;

    // load this wave's 16 tokens of z (f32, 64 ch each) into zb0
    {
        const float4* xg = (const float4*)x + (size_t)waveTok * (DD / 4);
        #pragma unroll
        for (int i = 0; i < 8; ++i) {
            int e  = lane + i * 32;          // 0..255 float4s
            int tk = e >> 4;
            int c4 = e & 15;
            float4 v = xg[tk * (DD / 4) + c4];
            *(float4*)(zb0 + tk * ZR + c4 * 4) = v;
        }
    }

    float* zc = zb0;
    float* zn = zb1;

    for (int li = LL - 1; li >= 0; --li) {
        __syncthreads();   // prior layer done with weights; epilogue LDS writes visible

        // ---- stage layer weights (f32 -> f16, [n][k] transposed) ----
        {
            const float4* g1 = (const float4*)(W1 + (size_t)li * HALFD * HH);
            #pragma unroll 4
            for (int i = 0; i < 16; ++i) {               // 2048 float4 / 128 thr
                int e = threadIdx.x + i * THREADS;
                int k = e >> 6;
                int n = (e & 63) << 2;
                float4 v = g1[e];
                W1T[(n + 0) * W1R + k] = (_Float16)v.x;
                W1T[(n + 1) * W1R + k] = (_Float16)v.y;
                W1T[(n + 2) * W1R + k] = (_Float16)v.z;
                W1T[(n + 3) * W1R + k] = (_Float16)v.w;
            }
            const float4* g2 = (const float4*)(W2 + (size_t)li * HH * HH);
            for (int i = 0; i < 128; ++i) {              // 16384 float4 / 128 thr
                int e = threadIdx.x + i * THREADS;
                int k = e >> 6;
                int n = (e & 63) << 2;
                float4 v = g2[e];
                W2T[(n + 0) * W2R + k] = (_Float16)v.x;
                W2T[(n + 1) * W2R + k] = (_Float16)v.y;
                W2T[(n + 2) * W2R + k] = (_Float16)v.z;
                W2T[(n + 3) * W2R + k] = (_Float16)v.w;
            }
            const float4* g3 = (const float4*)(W3 + (size_t)li * HH * DD);
            #pragma unroll 4
            for (int i = 0; i < 32; ++i) {               // 4096 float4 / 128 thr
                int e = threadIdx.x + i * THREADS;
                int k = e >> 4;
                int n = (e & 15) << 2;
                float4 v = g3[e];
                W3T[(n + 0) * W3R + k] = (_Float16)v.x;
                W3T[(n + 1) * W3R + k] = (_Float16)v.y;
                W3T[(n + 2) * W3R + k] = (_Float16)v.z;
                W3T[(n + 3) * W3R + k] = (_Float16)v.w;
            }
            for (int i = threadIdx.x; i < HH; i += THREADS) {
                b1s[i] = b1[li * HH + i];
                b2s[i] = b2[li * HH + i];
            }
            if (threadIdx.x < DD) b3s[threadIdx.x] = b3[li * DD + threadIdx.x];
        }

        // prefetch next layer's weights into the cache hierarchy (overlaps compute)
        if (li > 0) {
            int nli = li - 1;
            const char* p2 = (const char*)(W2 + (size_t)nli * HH * HH);
            #pragma unroll 4
            for (int i = 0; i < 16; ++i)                 // 256KB / 128thr / 128B lines
                __builtin_prefetch(p2 + (threadIdx.x + i * THREADS) * 128, 0, 1);
            const char* p1 = (const char*)(W1 + (size_t)nli * HALFD * HH);
            __builtin_prefetch(p1 + threadIdx.x * 128, 0, 1);          // covers 16KB of the 32KB W1 slice
            const char* p3 = (const char*)(W3 + (size_t)nli * HH * DD);
            #pragma unroll
            for (int i = 0; i < 4; ++i)
                __builtin_prefetch(p3 + (threadIdx.x + i * THREADS) * 128, 0, 1);
        }
        __syncthreads();

        // channel parity: li odd -> id = even channels; li even -> id = odd
        const int p_id = (li & 1) ? 0 : 1;
        const int p_s  = 1 - p_id;

        // ---- GEMM1: z_id (16x32) @ W1 (32x256) -> h1, leaky-relu ----
        v16h a1;
        #pragma unroll
        for (int j = 0; j < 16; ++j) {
            int k = hi * 16 + j;
            a1[j] = (_Float16)zc[m * ZR + 2 * k + p_id];
        }
        for (int nt = 0; nt < 16; ++nt) {
            int n = nt * 16 + m;
            v16h bfr = ld_frag(&W1T[n * W1R + hi * 16], &W1T[n * W1R + hi * 16 + 8]);
            float bias = b1s[n];
            v8f c;
            #pragma unroll
            for (int r = 0; r < 8; ++r) c[r] = bias;
            c = __builtin_amdgcn_wmma_f32_16x16x32_f16(false, a1, false, bfr,
                                                       (short)0, c, false, false);
            #pragma unroll
            for (int r = 0; r < 8; ++r)
                h1[(r + hi * 8) * HH + n] = (_Float16)lrelu(c[r]);
        }

        // ---- GEMM2: h1 (16x256) @ W2 (256x256) -> h2, leaky-relu ----
        // A-fragments shared across all 16 N-tiles: hoist (8 x v16h = 64 VGPRs)
        v16h a2[8];
        #pragma unroll
        for (int kt = 0; kt < 8; ++kt)
            a2[kt] = ld_frag(&h1[m * HH + kt * 32 + hi * 16],
                             &h1[m * HH + kt * 32 + hi * 16 + 8]);
        for (int nt = 0; nt < 16; ++nt) {
            int n = nt * 16 + m;
            float bias = b2s[n];
            v8f c;
            #pragma unroll
            for (int r = 0; r < 8; ++r) c[r] = bias;
            #pragma unroll
            for (int kt = 0; kt < 8; ++kt) {
                v16h b = ld_frag(&W2T[n * W2R + kt * 32 + hi * 16],
                                 &W2T[n * W2R + kt * 32 + hi * 16 + 8]);
                c = __builtin_amdgcn_wmma_f32_16x16x32_f16(false, a2[kt], false, b,
                                                           (short)0, c, false, false);
            }
            #pragma unroll
            for (int r = 0; r < 8; ++r)
                h2[(r + hi * 8) * HH + n] = (_Float16)lrelu(c[r]);
        }

        // ---- GEMM3: h2 (16x256) @ W3 (256x64) -> st (4 reg tiles) ----
        v16h a3[8];
        #pragma unroll
        for (int kt = 0; kt < 8; ++kt)
            a3[kt] = ld_frag(&h2[m * HH + kt * 32 + hi * 16],
                             &h2[m * HH + kt * 32 + hi * 16 + 8]);
        v8f st[4];
        #pragma unroll
        for (int nt = 0; nt < 4; ++nt) {
            int n = nt * 16 + m;
            float bias = b3s[n];
            v8f c;
            #pragma unroll
            for (int r = 0; r < 8; ++r) c[r] = bias;
            #pragma unroll
            for (int kt = 0; kt < 8; ++kt) {
                v16h b = ld_frag(&W3T[n * W3R + kt * 32 + hi * 16],
                                 &W3T[n * W3R + kt * 32 + hi * 16 + 8]);
                c = __builtin_amdgcn_wmma_f32_16x16x32_f16(false, a3[kt], false, b,
                                                           (short)0, c, false, false);
            }
            st[nt] = c;
        }

        // ---- epilogue: z_id copy, z_s = (z_s + t)*exp(s), logdet partials ----
        #pragma unroll 4
        for (int tok = 0; tok < TOK_PER_WAVE; ++tok)
            zn[tok * ZR + lane] = zc[tok * ZR + (lane << 1) + p_id];

        #pragma unroll
        for (int pp = 0; pp < 2; ++pp) {
            int scol = pp * 16 + m;          // 0..31
            float wr = wres[scol];
            #pragma unroll
            for (int r = 0; r < 8; ++r) {
                int tok = r + hi * 8;
                float s  = wr * tanhf(st[pp][r]);
                float t  = st[pp + 2][r];
                float zs = zc[tok * ZR + 2 * scol + p_s];
                zn[tok * ZR + HALFD + scol] = (zs + t) * __expf(s);
                ldj_part[r] += s;            // register-only accumulation
            }
        }

        float* tmp = zc; zc = zn; zn = tmp;
    }

    // ---- write outputs: z (BN x 64) then log_det_J (BN) ----
    {
        float4* og = (float4*)out + (size_t)waveTok * (DD / 4);
        #pragma unroll
        for (int i = 0; i < 8; ++i) {
            int e  = lane + i * 32;
            int tk = e >> 4;
            int c4 = e & 15;
            og[tk * (DD / 4) + c4] = *(float4*)(zc + tk * ZR + c4 * 4);
        }
    }
    // cross-lane reduce logdet partials over the 16 columns in each half-wave
    #pragma unroll
    for (int off = 1; off < 16; off <<= 1) {
        #pragma unroll
        for (int r = 0; r < 8; ++r)
            ldj_part[r] += __shfl_xor(ldj_part[r], off, 32);
    }
    if (m == 0) {
        float* lo = out + (size_t)BN * DD + waveTok + hi * 8;
        #pragma unroll
        for (int r = 0; r < 8; ++r) lo[r] = ldj_part[r];
    }
}

extern "C" void kernel_launch(void* const* d_in, const int* in_sizes, int n_in,
                              void* d_out, int out_size, void* d_ws, size_t ws_size,
                              hipStream_t stream) {
    (void)in_sizes; (void)n_in; (void)out_size; (void)d_ws; (void)ws_size;
    const float* x  = (const float*)d_in[0];
    // d_in[1] = mask (int32) — parity schedule is fixed, folded into the kernel
    const float* W1 = (const float*)d_in[2];
    const float* b1 = (const float*)d_in[3];
    const float* W2 = (const float*)d_in[4];
    const float* b2 = (const float*)d_in[5];
    const float* W3 = (const float*)d_in[6];
    const float* b3 = (const float*)d_in[7];
    const float* rv = (const float*)d_in[8];
    const float* rg = (const float*)d_in[9];
    float* out = (float*)d_out;

    hipFuncSetAttribute(reinterpret_cast<const void*>(realnvp_fused),
                        hipFuncAttributeMaxDynamicSharedMemorySize, SMEM_BYTES);

    dim3 grid(BN / TOK_PER_BLOCK);   // 4096 workgroups
    dim3 block(THREADS);             // 128 threads = 4 wave32
    realnvp_fused<<<grid, block, SMEM_BYTES, stream>>>(x, W1, b1, W2, b2, W3, b3,
                                                       rv, rg, out);
}
